// MultiHeadAttention_6906307412547
// MI455X (gfx1250) — matrix operand order repaired and checked by reading, not compiled, not run on hardware
//
#include <hip/hip_runtime.h>

// ---------------------------------------------------------------------------
// MI455X (gfx1250) CDNA5 multi-head attention, bf16 WMMA (f32 accumulate),
// double-buffered LDS staging via GLOBAL_LOAD_ASYNC_TO_LDS_B128 (ASYNCcnt).
// B=4, S=2048, D=1024, H=16, HD=64.
// ---------------------------------------------------------------------------

typedef __bf16 bf16;
typedef bf16          v16bf __attribute__((ext_vector_type(16)));
typedef float         v8f   __attribute__((ext_vector_type(8)));
typedef unsigned int  u32x4 __attribute__((ext_vector_type(4)));
typedef unsigned int  u32x2 __attribute__((ext_vector_type(2)));
typedef float         f32x4 __attribute__((ext_vector_type(4)));

#define DEV static __device__ __forceinline__

DEV unsigned short f2bf(float f) {
  union { float f; unsigned int u; } v; v.f = f;
  unsigned int u = v.u;
  u += 0x7FFFu + ((u >> 16) & 1u);   // round-to-nearest-even
  return (unsigned short)(u >> 16);
}

DEV v8f wmma_bf16(v16bf a, v16bf b, v8f c) {
  return __builtin_amdgcn_wmma_f32_16x16x32_bf16(false, a, false, b, (short)0, c,
                                                 false, false);
}

DEV v16bf load_frag(const unsigned short* p0, const unsigned short* p1) {
  union { u32x4 q[2]; v16bf v; } f;
  f.q[0] = *(const u32x4*)p0;
  f.q[1] = *(const u32x4*)p1;
  return f.v;
}

// Async 16-byte global -> LDS copy (per lane). Tracked by ASYNCcnt.
DEV void async_b128(const unsigned short* gsrc, unsigned short* ldst) {
  unsigned l = (unsigned)(unsigned long long)ldst;  // low 32 bits = LDS address
  asm volatile("global_load_async_to_lds_b128 %0, %1, off"
               :: "v"(l), "v"(gsrc) : "memory");
}
DEV void wait_async0() { asm volatile("s_wait_asynccnt 0" ::: "memory"); }

// ---------------------------------------------------------------------------
// f32 -> bf16 bulk conversion (4 elements / thread)
// ---------------------------------------------------------------------------
__global__ __launch_bounds__(256) void k_f32_to_bf16(const float* __restrict__ src,
                                                     unsigned short* __restrict__ dst) {
  size_t i = ((size_t)blockIdx.x * 256 + threadIdx.x) * 4;
  f32x4 d = *(const f32x4*)(src + i);
  u32x2 o;
  o.x = (unsigned)f2bf(d.x) | ((unsigned)f2bf(d.y) << 16);
  o.y = (unsigned)f2bf(d.z) | ((unsigned)f2bf(d.w) << 16);
  *(u32x2*)(dst + i) = o;
}

// ---------------------------------------------------------------------------
// Shared GEMM core: block tile 128x64, 8 waves (4x2), 32x32 per wave,
// K-step 64, ping-pong LDS, async A staging, register-prefetched B staging
// (B stored transposed in LDS so B fragments are contiguous 32B reads).
//   waddr(kRow, col) -> pointer into W for row kRow, column offset col
//   epi(acc, mBase, wM, wN, hl, l16) -> writes results
// ---------------------------------------------------------------------------
template <typename WAddrF, typename EpiF>
DEV void gemm_core(const unsigned short* __restrict__ X,
                   unsigned short* Al, unsigned short* Bl,
                   WAddrF waddr, EpiF epi) {
  constexpr int D = 1024;
  const int t = threadIdx.x;
  const int lane = t & 31, w = t >> 5;
  const int wM = w & 3, wN = w >> 2;
  const int hl = lane >> 4, l16 = lane & 15;
  const int mBase = blockIdx.x * 128;

  v8f acc[2][2] = {};

  const int arow = t >> 1, ap = t & 1;    // A: 2 thr/row, 32 elems each
  const int bkk = t >> 2, bnp = t & 3;    // B: 16 elems, transposed write

  auto stageA = [&](int buf, int k0) {
    const unsigned short* src = X + (size_t)(mBase + arow) * D + k0 + ap * 32;
    unsigned short* dst = Al + buf * (128 * 72) + arow * 72 + ap * 32;
#pragma unroll
    for (int j = 0; j < 4; ++j) async_b128(src + j * 8, dst + j * 8);
  };

  { // prologue: stage buffer 0
    stageA(0, 0);
    const unsigned short* src = waddr(bkk, bnp * 16);
    union { u32x4 q[2]; unsigned short s[16]; } br;
    br.q[0] = *(const u32x4*)src;
    br.q[1] = *(const u32x4*)(src + 8);
#pragma unroll
    for (int j = 0; j < 16; ++j) Bl[(bnp * 16 + j) * 72 + bkk] = br.s[j];
    wait_async0();
  }
  __syncthreads();

  for (int i = 0; i < 16; ++i) {
    const int cur = i & 1, nxt = cur ^ 1;
    const bool pre = (i < 15);
    union { u32x4 q[2]; unsigned short s[16]; } br;
    if (pre) {
      stageA(nxt, (i + 1) * 64);
      const unsigned short* src = waddr((i + 1) * 64 + bkk, bnp * 16);
      br.q[0] = *(const u32x4*)src;
      br.q[1] = *(const u32x4*)(src + 8);
    }
    const unsigned short* Ab = Al + cur * (128 * 72);
    const unsigned short* Bb = Bl + cur * (64 * 72);
#pragma unroll
    for (int m2 = 0; m2 < 2; ++m2) {
      const unsigned short* ar = Ab + (wM * 32 + m2 * 16 + l16) * 72;
      v16bf af0 = load_frag(ar + 16 * hl, ar + 16 * hl + 8);
      v16bf af1 = load_frag(ar + 32 + 16 * hl, ar + 32 + 16 * hl + 8);
#pragma unroll
      for (int n2 = 0; n2 < 2; ++n2) {
        const unsigned short* brd = Bb + (wN * 32 + n2 * 16 + l16) * 72;
        v16bf bf0 = load_frag(brd + 16 * hl, brd + 16 * hl + 8);
        v16bf bf1 = load_frag(brd + 32 + 16 * hl, brd + 32 + 16 * hl + 8);
        acc[m2][n2] = wmma_bf16(af0, bf0, acc[m2][n2]);
        acc[m2][n2] = wmma_bf16(af1, bf1, acc[m2][n2]);
      }
    }
    if (pre) {
      unsigned short* Bd = Bl + nxt * (64 * 72);
#pragma unroll
      for (int j = 0; j < 16; ++j) Bd[(bnp * 16 + j) * 72 + bkk] = br.s[j];
      wait_async0();
    }
    __syncthreads();
  }

  epi(acc, mBase, wM, wN, hl, l16);
}

// QKV projection: X[BS,1024] x W[h][1024][64] + bias[h][64] -> [b][h][s][64] bf16
__global__ __launch_bounds__(256) void k_gemm_qkv(
    const unsigned short* __restrict__ X, const unsigned short* __restrict__ W,
    const float* __restrict__ bias, unsigned short* __restrict__ O) {
  __shared__ __align__(16) unsigned short Al[2 * 128 * 72];
  __shared__ __align__(16) unsigned short Bl[2 * 64 * 72];
  const int head = blockIdx.y;

  auto waddr = [&](int kRow, int col) {
    return W + ((size_t)head * 1024 + kRow) * 64 + col;
  };
  auto epi = [&](const v8f (&acc)[2][2], int mBase, int wM, int wN, int hl,
                 int l16) {
#pragma unroll
    for (int m2 = 0; m2 < 2; ++m2) {
#pragma unroll
      for (int n2 = 0; n2 < 2; ++n2) {
#pragma unroll
        for (int v = 0; v < 8; ++v) {
          int m = mBase + wM * 32 + m2 * 16 + v + 8 * hl;
          int e = wN * 32 + n2 * 16 + l16;
          float val = acc[m2][n2][v] + bias[head * 64 + e];
          int b = m >> 11;
          int s = m & 2047;
          O[(((size_t)(b * 16 + head) * 2048) + s) * 64 + e] = f2bf(val);
        }
      }
    }
  };
  gemm_core(X, Al, Bl, waddr, epi);
}

// Output projection: Z[BS,1024] x Wo[1024,1024] + bo -> out f32 [BS,1024]
__global__ __launch_bounds__(256) void k_gemm_out(
    const unsigned short* __restrict__ X, const unsigned short* __restrict__ W,
    const float* __restrict__ bias, float* __restrict__ O) {
  __shared__ __align__(16) unsigned short Al[2 * 128 * 72];
  __shared__ __align__(16) unsigned short Bl[2 * 64 * 72];
  const int nBase = blockIdx.y * 64;

  auto waddr = [&](int kRow, int col) {
    return W + (size_t)kRow * 1024 + nBase + col;
  };
  auto epi = [&](const v8f (&acc)[2][2], int mBase, int wM, int wN, int hl,
                 int l16) {
#pragma unroll
    for (int m2 = 0; m2 < 2; ++m2) {
#pragma unroll
      for (int n2 = 0; n2 < 2; ++n2) {
#pragma unroll
        for (int v = 0; v < 8; ++v) {
          int m = mBase + wM * 32 + m2 * 16 + v + 8 * hl;
          int n = nBase + wN * 32 + n2 * 16 + l16;
          O[(size_t)m * 1024 + n] = acc[m2][n2][v] + bias[n];
        }
      }
    }
  };
  gemm_core(X, Al, Bl, waddr, epi);
}

// ---------------------------------------------------------------------------
// Streaming-softmax attention. One block = one (b,h) x 128 query rows.
// 8 waves; wave w owns 16 query rows. Key chunks of 64, double-buffered:
// K staged via async-to-LDS, V prefetched to regs and scatter-transposed.
// Masked scores are set to 0.0 and PARTICIPATE in softmax (faithful).
// ---------------------------------------------------------------------------
__global__ __launch_bounds__(256) void k_attn(
    const unsigned short* __restrict__ Q, const unsigned short* __restrict__ K,
    const unsigned short* __restrict__ V, const int* __restrict__ mask,
    unsigned short* __restrict__ Z) {
  constexpr int S = 2048;
  __shared__ __align__(16) unsigned short Ql[128 * 72];      // [qrow][e]
  __shared__ __align__(16) unsigned short Kl[2 * 64 * 72];   // [key][e]
  __shared__ __align__(16) unsigned short Vt[2 * 64 * 72];   // [e][key]
  __shared__ __align__(16) unsigned short Pl[8 * 16 * 72];   // per-wave P

  const int t = threadIdx.x;
  const int lane = t & 31, w = t >> 5;
  const int hl = lane >> 4, l16 = lane & 15;
  const int bh = blockIdx.x >> 4;
  const int qTile = blockIdx.x & 15;
  const int b = bh >> 4, h = bh & 15;
  const int qBase = qTile * 128;
  const size_t headBase = ((size_t)(b * 16 + h)) * S * 64;

  const int krow = t >> 2, kp = t & 3;
  const int vkk = t & 63, veb = t >> 6;

  auto stageK = [&](int buf, int chunk) {
    const unsigned short* src =
        K + headBase + (size_t)(chunk * 64 + krow) * 64 + kp * 16;
    unsigned short* dst = Kl + buf * (64 * 72) + krow * 72 + kp * 16;
    async_b128(src, dst);
    async_b128(src + 8, dst + 8);
  };

  { // stage Q tile (async) + first K chunk (async) + first V chunk (sync)
    int row = t >> 1, part = t & 1;
    const unsigned short* src =
        Q + headBase + (size_t)(qBase + row) * 64 + part * 32;
    unsigned short* dst = Ql + row * 72 + part * 32;
#pragma unroll
    for (int j = 0; j < 4; ++j) async_b128(src + j * 8, dst + j * 8);
    stageK(0, 0);
    const unsigned short* vs = V + headBase + (size_t)vkk * 64 + veb * 16;
    union { u32x4 q[2]; unsigned short s[16]; } vv;
    vv.q[0] = *(const u32x4*)vs;
    vv.q[1] = *(const u32x4*)(vs + 8);
#pragma unroll
    for (int j = 0; j < 16; ++j) Vt[(veb * 16 + j) * 72 + vkk] = vv.s[j];
    wait_async0();
  }
  __syncthreads();

  v16bf qf[2];
  {
    const unsigned short* qr = Ql + (w * 16 + l16) * 72;
#pragma unroll
    for (int ks = 0; ks < 2; ++ks)
      qf[ks] = load_frag(qr + ks * 32 + 16 * hl, qr + ks * 32 + 16 * hl + 8);
  }

  v8f oacc[4] = {};
  float rowM[8], rowL[8];
#pragma unroll
  for (int v = 0; v < 8; ++v) { rowM[v] = -3.0e38f; rowL[v] = 0.f; }

  unsigned short* Pw = Pl + w * (16 * 72);

  for (int c = 0; c < 32; ++c) {
    const int cur = c & 1, nxt = cur ^ 1;
    const bool pre = (c < 31);
    const int kBase = c * 64;

    union { u32x4 q[2]; unsigned short s[16]; } vv;
    if (pre) { // prefetch next chunk while computing on this one
      stageK(nxt, c + 1);
      const unsigned short* vs =
          V + headBase + (size_t)((c + 1) * 64 + vkk) * 64 + veb * 16;
      vv.q[0] = *(const u32x4*)vs;
      vv.q[1] = *(const u32x4*)(vs + 8);
    }

    const unsigned short* Kb = Kl + cur * (64 * 72);
    const unsigned short* Vb = Vt + cur * (64 * 72);

    // --- scores: Q * K^T ---
    v8f sacc[4] = {};
#pragma unroll
    for (int nt = 0; nt < 4; ++nt) {
      const unsigned short* kr = Kb + (nt * 16 + l16) * 72;
#pragma unroll
      for (int ks = 0; ks < 2; ++ks) {
        v16bf kf = load_frag(kr + ks * 32 + 16 * hl, kr + ks * 32 + 16 * hl + 8);
        sacc[nt] = wmma_bf16(qf[ks], kf, sacc[nt]);
      }
    }

    // --- scale, mask-to-zero, online softmax ---
    float corr[8];
#pragma unroll
    for (int v = 0; v < 8; ++v) {
      const int sq = qBase + w * 16 + v + 8 * hl;
      const int* mrow = mask + ((size_t)b * S + sq) * S + kBase + l16;
#pragma unroll
      for (int nt = 0; nt < 4; ++nt) {
        float sv = sacc[nt][v] * 0.125f;
        if (mrow[nt * 16] == 0) sv = 0.f;
        sacc[nt][v] = sv;
      }
      float cm = fmaxf(fmaxf(sacc[0][v], sacc[1][v]), fmaxf(sacc[2][v], sacc[3][v]));
      cm = fmaxf(cm, __shfl_xor(cm, 1, 32));
      cm = fmaxf(cm, __shfl_xor(cm, 2, 32));
      cm = fmaxf(cm, __shfl_xor(cm, 4, 32));
      cm = fmaxf(cm, __shfl_xor(cm, 8, 32));
      float nm = fmaxf(rowM[v], cm);
      corr[v] = __expf(rowM[v] - nm);
      rowM[v] = nm;
      float ps = 0.f;
#pragma unroll
      for (int nt = 0; nt < 4; ++nt) {
        float p = __expf(sacc[nt][v] - nm);
        sacc[nt][v] = p;
        ps += p;
      }
      ps += __shfl_xor(ps, 1, 32);
      ps += __shfl_xor(ps, 2, 32);
      ps += __shfl_xor(ps, 4, 32);
      ps += __shfl_xor(ps, 8, 32);
      rowL[v] = rowL[v] * corr[v] + ps;
#pragma unroll
      for (int nt = 0; nt < 4; ++nt)    // C-layout -> A-layout via LDS
        Pw[(v + 8 * hl) * 72 + nt * 16 + l16] = f2bf(sacc[nt][v]);
    }

#pragma unroll
    for (int et = 0; et < 4; ++et) {
#pragma unroll
      for (int v = 0; v < 8; ++v) oacc[et][v] *= corr[v];
    }

    asm volatile("s_wait_dscnt 0" ::: "memory");  // P stores -> our P loads

    // --- O += P * V ---
    v16bf pf[2];
    {
      const unsigned short* pr = Pw + l16 * 72;
#pragma unroll
      for (int ks = 0; ks < 2; ++ks)
        pf[ks] = load_frag(pr + ks * 32 + 16 * hl, pr + ks * 32 + 16 * hl + 8);
    }
#pragma unroll
    for (int et = 0; et < 4; ++et) {
      const unsigned short* vr = Vb + (et * 16 + l16) * 72;
#pragma unroll
      for (int ks = 0; ks < 2; ++ks) {
        v16bf vf = load_frag(vr + ks * 32 + 16 * hl, vr + ks * 32 + 16 * hl + 8);
        oacc[et] = wmma_bf16(pf[ks], vf, oacc[et]);
      }
    }

    if (pre) { // finish staging next buffer, drain our async ops
      unsigned short* Vd = Vt + nxt * (64 * 72);
#pragma unroll
      for (int j = 0; j < 16; ++j) Vd[(veb * 16 + j) * 72 + vkk] = vv.s[j];
      wait_async0();
    }
    __syncthreads();
  }

#pragma unroll
  for (int v = 0; v < 8; ++v) {
    float inv = 1.0f / rowL[v];
    const int sq = qBase + w * 16 + v + 8 * hl;
#pragma unroll
    for (int et = 0; et < 4; ++et) {
      int e = et * 16 + l16;
      Z[((size_t)b * S + sq) * 1024 + h * 64 + e] = f2bf(oacc[et][v] * inv);
    }
  }
}

// ---------------------------------------------------------------------------
extern "C" void kernel_launch(void* const* d_in, const int* in_sizes, int n_in,
                              void* d_out, int out_size, void* d_ws, size_t ws_size,
                              hipStream_t stream) {
  (void)in_sizes; (void)n_in; (void)out_size; (void)ws_size;

  const float* x_v  = (const float*)d_in[0];
  const float* x_k  = (const float*)d_in[1];
  const float* x_q  = (const float*)d_in[2];
  const int*   mask = (const int*)d_in[3];
  const float* Wq   = (const float*)d_in[4];
  const float* bq   = (const float*)d_in[5];
  const float* Wk   = (const float*)d_in[6];
  const float* bk   = (const float*)d_in[7];
  const float* Wv   = (const float*)d_in[8];
  const float* bv   = (const float*)d_in[9];
  const float* Wo   = (const float*)d_in[10];
  const float* bo   = (const float*)d_in[11];
  float* out = (float*)d_out;

  char* ws = (char*)d_ws;
  const size_t SZ_X = (size_t)4 * 2048 * 1024 * 2;
  const size_t SZ_W = (size_t)16 * 1024 * 64 * 2;
  unsigned short* xq_bf = (unsigned short*)(ws);
  unsigned short* xk_bf = (unsigned short*)(ws + SZ_X);
  unsigned short* xv_bf = (unsigned short*)(ws + 2 * SZ_X);
  unsigned short* wq_bf = (unsigned short*)(ws + 3 * SZ_X);
  unsigned short* wk_bf = (unsigned short*)(ws + 3 * SZ_X + SZ_W);
  unsigned short* wv_bf = (unsigned short*)(ws + 3 * SZ_X + 2 * SZ_W);
  unsigned short* wo_bf = (unsigned short*)(ws + 3 * SZ_X + 3 * SZ_W);
  unsigned short* qb    = (unsigned short*)(ws + 3 * SZ_X + 4 * SZ_W);
  unsigned short* kb    = (unsigned short*)(ws + 4 * SZ_X + 4 * SZ_W);
  unsigned short* vb    = (unsigned short*)(ws + 5 * SZ_X + 4 * SZ_W);
  unsigned short* zb    = (unsigned short*)(ws + 6 * SZ_X + 4 * SZ_W);

  const int NX = 4 * 2048 * 1024;
  const int NW = 16 * 1024 * 64;
  k_f32_to_bf16<<<NX / 1024, 256, 0, stream>>>(x_q, xq_bf);
  k_f32_to_bf16<<<NX / 1024, 256, 0, stream>>>(x_k, xk_bf);
  k_f32_to_bf16<<<NX / 1024, 256, 0, stream>>>(x_v, xv_bf);
  k_f32_to_bf16<<<NW / 1024, 256, 0, stream>>>(Wq, wq_bf);
  k_f32_to_bf16<<<NW / 1024, 256, 0, stream>>>(Wk, wk_bf);
  k_f32_to_bf16<<<NW / 1024, 256, 0, stream>>>(Wv, wv_bf);
  k_f32_to_bf16<<<NW / 1024, 256, 0, stream>>>(Wo, wo_bf);

  dim3 gq(64, 16);
  k_gemm_qkv<<<gq, 256, 0, stream>>>(xq_bf, wq_bf, bq, qb);
  k_gemm_qkv<<<gq, 256, 0, stream>>>(xk_bf, wk_bf, bk, kb);
  k_gemm_qkv<<<gq, 256, 0, stream>>>(xv_bf, wv_bf, bv, vb);

  k_attn<<<1024, 256, 0, stream>>>(qb, kb, vb, mask, zb);

  k_gemm_out<<<gq, 256, 0, stream>>>(zb, wo_bf, bo, out);
}
